// GnnProc_72206990181062
// MI455X (gfx1250) — matrix operand order repaired and checked by reading, not compiled, not run on hardware
//
#include <hip/hip_runtime.h>
#include <cstddef>

// ---------------------------------------------------------------------------
// MI455X / gfx1250 fused kernel for:
//   m1z = z@M1^T + b1 ; m2z = z@M2^T + b2 ; pooled = max_k m2z
//   m = relu(m1z + pooled) ; out = relu([z,m]@U^T + bU)
// Strategy: f16 WMMA (16x16x32) with fp32 accumulate; weights in LDS;
// two passes over z (max-pool dependency), second pass L2-resident.
// ---------------------------------------------------------------------------

typedef __attribute__((ext_vector_type(16))) _Float16 v16h;
typedef __attribute__((ext_vector_type(8)))  _Float16 v8h;
typedef __attribute__((ext_vector_type(8)))  float    v8f;

#define ENC   128
#define HID   128
#define BATCH 32
#define SEQ   8192
#define WSTRIDE 144   // padded LDS row stride in halves: 288B (32B aligned, 8-bank rotate)

// ---- f32 CAS-based atomic max (pooled reduction across workgroups) --------
__device__ __forceinline__ void atomicMaxF(float* addr, float val) {
  unsigned int* ua = (unsigned int*)addr;
  unsigned int cur = __float_as_uint(*addr);
  while (__uint_as_float(cur) < val) {
    unsigned int old = atomicCAS(ua, cur, __float_as_uint(val));
    if (old == cur) break;
    cur = old;
  }
}

// ---- A-operand (16x32 f16) from a row of z (fp32), converting on the fly --
// 16-bit A layout: lanes 0-15: row M=lane, K = {0..7, 16..23};
//                  lanes 16-31: row M=lane-16, K = {8..15, 24..31}.
__device__ __forceinline__ v16h load_a_f32(const float* __restrict__ rowp, int lane) {
  const float* p = rowp + ((lane & 16) ? 8 : 0);
  float4 f0 = *(const float4*)(p);
  float4 f1 = *(const float4*)(p + 4);
  float4 g0 = *(const float4*)(p + 16);
  float4 g1 = *(const float4*)(p + 20);
  v16h a;
  a[0]=(_Float16)f0.x; a[1]=(_Float16)f0.y; a[2]=(_Float16)f0.z; a[3]=(_Float16)f0.w;
  a[4]=(_Float16)f1.x; a[5]=(_Float16)f1.y; a[6]=(_Float16)f1.z; a[7]=(_Float16)f1.w;
  a[8]=(_Float16)g0.x; a[9]=(_Float16)g0.y; a[10]=(_Float16)g0.z; a[11]=(_Float16)g0.w;
  a[12]=(_Float16)g1.x; a[13]=(_Float16)g1.y; a[14]=(_Float16)g1.z; a[15]=(_Float16)g1.w;
  return a;
}

// ---- B-operand (32x16 f16) from LDS weight matrix W[f][e] (row-major) -----
// B layout: lanes 0-15: col N=lane, K = {0..7, 16..23}; lanes 16-31: K = {8..15, 24..31}.
__device__ __forceinline__ v16h load_b_lds(const _Float16* w, int ct, int kc, int lane) {
  int f = ct * 16 + (lane & 15);
  int e = kc * 32 + ((lane & 16) ? 8 : 0);
  v8h lo = *(const v8h*)(w + f * WSTRIDE + e);
  v8h hi = *(const v8h*)(w + f * WSTRIDE + e + 16);
  v16h r;
#pragma unroll
  for (int i = 0; i < 8; ++i) { r[i] = lo[i]; r[8 + i] = hi[i]; }
  return r;
}

__device__ __forceinline__ v8f wmma_f16(v16h a, v16h b, v8f c) {
  return __builtin_amdgcn_wmma_f32_16x16x32_f16(false, a, false, b, (short)0, c, false, false);
}

// ---------------------------------------------------------------------------
__global__ void k_init(float* __restrict__ pooled, int n) {
  int i = blockIdx.x * blockDim.x + threadIdx.x;
  if (i < n) pooled[i] = -3.4e38f;
}

// Phase 1: pooled[b][e] = max_k (z[b,k,:] . M2_w[e,:])   (bias added in phase 2)
__global__ __launch_bounds__(256) void k_pool(const float* __restrict__ z,
                                              const float* __restrict__ M2w,
                                              float* __restrict__ pooled) {
  extern __shared__ _Float16 sm[];                 // M2 f16: [128][WSTRIDE]
  const int tid = threadIdx.x;
  for (int i = tid; i < ENC * ENC; i += 256)
    sm[(i >> 7) * WSTRIDE + (i & 127)] = (_Float16)M2w[i];
  __syncthreads();

  const int lane = tid & 31, wave = tid >> 5;
  const int wg = blockIdx.x;                       // 256 WGs: 8 per batch
  const int b = wg >> 3;
  const int rowbase = (wg & 7) * 1024;

  float rm[8];
#pragma unroll
  for (int i = 0; i < 8; ++i) rm[i] = -3.4e38f;

  for (int t = 0; t < 8; ++t) {                    // 8 x 16-row tiles per wave
    const int r0 = rowbase + (t * 8 + wave) * 16;
    const float* rowp = z + ((size_t)b * SEQ + r0 + (lane & 15)) * ENC;
    v16h A[4];
#pragma unroll
    for (int kc = 0; kc < 4; ++kc) A[kc] = load_a_f32(rowp + kc * 32, lane);
#pragma unroll
    for (int ct = 0; ct < 8; ++ct) {
      v8f acc = {};
#pragma unroll
      for (int kc = 0; kc < 4; ++kc)
        acc = wmma_f16(A[kc], load_b_lds(sm, ct, kc, lane), acc);
      float m = acc[0];                            // per-lane: max over M rows 0..7 / 8..15
#pragma unroll
      for (int i = 1; i < 8; ++i) m = fmaxf(m, acc[i]);
      rm[ct] = fmaxf(rm[ct], m);
    }
  }
#pragma unroll
  for (int ct = 0; ct < 8; ++ct) {                 // merge lanes L / L+16 (M halves)
    float v = fmaxf(rm[ct], __shfl_xor(rm[ct], 16, 32));
    if (lane < 16) atomicMaxF(&pooled[b * ENC + ct * 16 + lane], v);
  }
}

// Phase 2: out = relu(z@Uz^T + relu(z@M1^T + b1 + pooled + b2)@Um^T + bU)
__global__ __launch_bounds__(256) void k_main(const float* __restrict__ z,
                                              const float* __restrict__ M1w,
                                              const float* __restrict__ M1b,
                                              const float* __restrict__ M2b,
                                              const float* __restrict__ Uw,
                                              const float* __restrict__ Ub,
                                              const float* __restrict__ pooled,
                                              float* __restrict__ out) {
  extern __shared__ _Float16 sm[];
  _Float16* sM1 = sm;                              // [128][WSTRIDE]
  _Float16* sUz = sM1 + ENC * WSTRIDE;             // U_w[:, :128]
  _Float16* sUm = sUz + ENC * WSTRIDE;             // U_w[:, 128:]
  _Float16* sMm = sUm + ENC * WSTRIDE;             // per-wave m tiles [8][16][WSTRIDE]

  const int tid = threadIdx.x;
  for (int i = tid; i < ENC * ENC; i += 256) {
    const int h = i >> 7, e = i & 127;
    sM1[h * WSTRIDE + e] = (_Float16)M1w[i];
    sUz[h * WSTRIDE + e] = (_Float16)Uw[h * 2 * ENC + e];
    sUm[h * WSTRIDE + e] = (_Float16)Uw[h * 2 * ENC + ENC + e];
  }
  __syncthreads();

  const int lane = tid & 31, wave = tid >> 5;
  _Float16* mTile = sMm + wave * 16 * WSTRIDE;     // wave-private: no barrier needed
  const int wg = blockIdx.x;
  const int b = wg >> 3;
  const int rowbase = (wg & 7) * 1024;
  const int colL = lane & 15;
  const int rbase = (lane & 16) ? 8 : 0;

  for (int t = 0; t < 8; ++t) {
    const int r0 = rowbase + (t * 8 + wave) * 16;
    const size_t grow = (size_t)b * SEQ + r0;
    const float* rowp = z + (grow + (lane & 15)) * ENC;
    if (t < 7)                                     // global_prefetch_b8 for next tile
      __builtin_prefetch(rowp + 16 * 8 * ENC, 0, 1);

    v16h Az[4];
#pragma unroll
    for (int kc = 0; kc < 4; ++kc) Az[kc] = load_a_f32(rowp + kc * 32, lane);

    // GEMM 1: m = relu(z@M1^T + M1_b + pooled + M2_b), stored to LDS in A-layout rows
#pragma unroll
    for (int ct = 0; ct < 8; ++ct) {
      v8f acc = {};
#pragma unroll
      for (int kc = 0; kc < 4; ++kc)
        acc = wmma_f16(Az[kc], load_b_lds(sM1, ct, kc, lane), acc);
      const int col = ct * 16 + colL;
      const float bias = M1b[col] + pooled[b * ENC + col] + M2b[col];
#pragma unroll
      for (int v = 0; v < 8; ++v) {
        float mv = fmaxf(acc[v] + bias, 0.f);
        mTile[(rbase + v) * WSTRIDE + col] = (_Float16)mv;   // C-layout -> row-major
      }
    }
    asm volatile("s_wait_dscnt 0" ::: "memory");   // same-wave LDS RAW fence

    // Reload m as A-operand (transpose via LDS)
    v16h Am[4];
    const _Float16* mrow = mTile + (lane & 15) * WSTRIDE + ((lane & 16) ? 8 : 0);
#pragma unroll
    for (int kc = 0; kc < 4; ++kc) {
      v8h lo = *(const v8h*)(mrow + kc * 32);
      v8h hi = *(const v8h*)(mrow + kc * 32 + 16);
      v16h a;
#pragma unroll
      for (int i = 0; i < 8; ++i) { a[i] = lo[i]; a[8 + i] = hi[i]; }
      Am[kc] = a;
    }

    // GEMM 2: out = relu(z@Uz^T + m@Um^T + U_b)
#pragma unroll
    for (int ct = 0; ct < 8; ++ct) {
      v8f acc = {};
#pragma unroll
      for (int kc = 0; kc < 4; ++kc) {
        acc = wmma_f16(Az[kc], load_b_lds(sUz, ct, kc, lane), acc);
        acc = wmma_f16(Am[kc], load_b_lds(sUm, ct, kc, lane), acc);
      }
      const int col = ct * 16 + colL;
      const float bias = Ub[col];
      float* op = out + (grow + rbase) * HID + col;
#pragma unroll
      for (int v = 0; v < 8; ++v)
        op[(size_t)v * HID] = fmaxf(acc[v] + bias, 0.f);
    }
  }
}

// ---------------------------------------------------------------------------
extern "C" void kernel_launch(void* const* d_in, const int* in_sizes, int n_in,
                              void* d_out, int out_size, void* d_ws, size_t ws_size,
                              hipStream_t stream) {
  const float* z   = (const float*)d_in[0];
  const float* M1w = (const float*)d_in[1];
  const float* M1b = (const float*)d_in[2];
  const float* M2w = (const float*)d_in[3];
  const float* M2b = (const float*)d_in[4];
  const float* Uw  = (const float*)d_in[5];
  const float* Ub  = (const float*)d_in[6];
  float* out    = (float*)d_out;
  float* pooled = (float*)d_ws;                    // 32*128 floats = 16 KB

  const size_t ldsW  = (size_t)ENC * WSTRIDE * sizeof(_Float16);      // 36864 B
  const size_t lds2  = 3 * ldsW + (size_t)8 * 16 * WSTRIDE * sizeof(_Float16); // 147456 B
  hipFuncSetAttribute((const void*)k_main, hipFuncAttributeMaxDynamicSharedMemorySize, (int)lds2);
  hipFuncSetAttribute((const void*)k_pool, hipFuncAttributeMaxDynamicSharedMemorySize, (int)ldsW);

  const int nP = BATCH * ENC;
  k_init<<<(nP + 255) / 256, 256, 0, stream>>>(pooled, nP);
  k_pool<<<BATCH * 8, 256, ldsW, stream>>>(z, M2w, pooled);
  k_main<<<BATCH * 8, 256, lds2, stream>>>(z, M1w, M1b, M2b, Uw, Ub, pooled, out);
}
